// MoEBlock_51625506898335
// MI455X (gfx1250) — matrix operand order repaired and checked by reading, not compiled, not run on hardware
//
#include <hip/hip_runtime.h>
#include <hip/hip_bf16.h>
#include <math.h>

// ---------------- problem constants ----------------
#define BB    8
#define SS    512
#define CC    768
#define HH    12
#define EE    8
#define KTOP  2
#define MLPH  3072
#define HD    64
#define TT    (BB * SS)          // 4096 tokens
#define LN_EPS 1e-5f

typedef __attribute__((ext_vector_type(16))) __bf16 v16bf;
typedef __attribute__((ext_vector_type(8)))  float  v8f;

union FragBF { v16bf v; uint4 u[2]; };
union Pack8  { uint4 u; __bf16 h[8]; };
union Pack4  { uint2 u; __bf16 h[4]; };
union Pack2  { unsigned int u; __bf16 h[2]; };
union F4     { float4 v; float f[4]; };

// WMMA: D = A(16x32 bf16) * B(32x16 bf16) + C(16x16 f32)
__device__ __forceinline__ v8f wmma_bf16(v16bf a, v16bf b, v8f c) {
    return __builtin_amdgcn_wmma_f32_16x16x32_bf16(
        /*neg_a=*/false, a, /*neg_b=*/false, b,
        /*c_mod=*/(short)0, c, /*reuse_a=*/false, /*reuse_b=*/false);
}

// A-matrix fragment (16x32, 16-bit): lane<16 holds K=[0..7]∪[16..23],
// lane>=16 holds K=[8..15]∪[24..31] of row M=lane%16.  (ISA §7.12.2)
__device__ __forceinline__ v16bf load_a_frag(const __bf16* base, int stride, int lane, int koff) {
    const int r   = lane & 15;
    const int klo = koff + ((lane >= 16) ? 8 : 0);
    const __bf16* p = base + (size_t)r * stride;
    FragBF f;
    f.u[0] = *reinterpret_cast<const uint4*>(p + klo);
    f.u[1] = *reinterpret_cast<const uint4*>(p + klo + 16);
    return f.v;
}

// B-matrix fragment (32x16, 16-bit), LDS stored [n][k]: lane<16 holds
// K=[0..7]∪[16..23], lane>=16 holds K=[8..15]∪[24..31] of column N=lane%16.
__device__ __forceinline__ v16bf load_b_frag(const __bf16* base, int stride, int lane, int koff) {
    const int n   = lane & 15;
    const int klo = koff + ((lane >= 16) ? 8 : 0);
    const __bf16* p = base + (size_t)n * stride + klo;
    FragBF f;
    f.u[0] = *reinterpret_cast<const uint4*>(p);
    f.u[1] = *reinterpret_cast<const uint4*>(p + 16);
    return f.v;
}

// ---------------- init: out = x ; zero stats ----------------
__global__ void init_kernel(const float* __restrict__ x, float* __restrict__ out,
                            float* __restrict__ stats) {
    size_t i = (size_t)blockIdx.x * 256 + threadIdx.x;
    out[i] = x[i];                 // grid sized exactly TT*CC/256
    if (i < 16) stats[i] = 0.f;
}

// ---------------- router: softmax + top-2 gate ----------------
__global__ void router_kernel(const float* __restrict__ x, const float* __restrict__ et,
                              float* __restrict__ gate, float* __restrict__ stats) {
    const int lane = threadIdx.x & 31;
    const int wave = threadIdx.x >> 5;
    const int tok  = blockIdx.x * 8 + wave;
    const float* xr = x + (size_t)tok * CC;

    float logits[EE];
#pragma unroll
    for (int e = 0; e < EE; ++e) {
        const float* er = et + (size_t)e * CC;
        float acc = 0.f;
        for (int c = lane; c < CC; c += 32) acc += xr[c] * er[c];
#pragma unroll
        for (int off = 16; off > 0; off >>= 1) acc += __shfl_xor(acc, off, 32);
        logits[e] = acc * 0.03608439182435161f;   // 1/sqrt(768)
    }
    float mx = logits[0];
#pragma unroll
    for (int e = 1; e < EE; ++e) mx = fmaxf(mx, logits[e]);
    float probs[EE]; float sum = 0.f;
#pragma unroll
    for (int e = 0; e < EE; ++e) { probs[e] = __expf(logits[e] - mx); sum += probs[e]; }
    float inv = 1.f / sum;
#pragma unroll
    for (int e = 0; e < EE; ++e) probs[e] *= inv;

    int i1 = 0; float p1 = probs[0];
#pragma unroll
    for (int e = 1; e < EE; ++e) if (probs[e] > p1) { p1 = probs[e]; i1 = e; }
    int i2 = (i1 == 0) ? 1 : 0; float p2 = probs[i2];
#pragma unroll
    for (int e = 0; e < EE; ++e)
        if (e != i1 && probs[e] > p2) { p2 = probs[e]; i2 = e; }
    float denom = 1.f / (p1 + p2);

    if (lane == 0) {
#pragma unroll
        for (int e = 0; e < EE; ++e) {
            float g = (e == i1) ? p1 * denom : ((e == i2) ? p2 * denom : 0.f);
            gate[(size_t)tok * EE + e] = g;
            atomicAdd(&stats[8 + e], probs[e]);    // P_i accumulator
        }
        atomicAdd(&stats[i1], 1.f);                // counts
        atomicAdd(&stats[i2], 1.f);
    }
}

// ---------------- balance loss ----------------
__global__ void loss_kernel(const float* __restrict__ et, const float* __restrict__ stats,
                            float* __restrict__ out_loss) {
    __shared__ float red[256];
    const int tid = threadIdx.x;
    float ss = 0.f;
    for (int i = tid; i < EE * CC; i += 256) { float v = et[i]; ss += v * v; }
    red[tid] = ss; __syncthreads();
    for (int off = 128; off > 0; off >>= 1) {
        if (tid < off) red[tid] += red[tid + off];
        __syncthreads();
    }
    if (tid == 0) {
        float dot = 0.f;
#pragma unroll
        for (int e = 0; e < EE; ++e) {
            float f = stats[e]     / (float)(TT * KTOP);
            float P = stats[8 + e] / (float)TT;
            dot += f * P;
        }
        out_loss[0] = (float)EE * dot + 0.01f * sqrtf(red[0]);
    }
}

// ---------------- layernorm -> bf16 ----------------
__global__ void ln_bf16_kernel(const float* __restrict__ in, const float* __restrict__ g,
                               const float* __restrict__ b, __bf16* __restrict__ out) {
    __shared__ float red[256];
    const int row = blockIdx.x, tid = threadIdx.x;
    const float* src = in + (size_t)row * CC;
    float v[3];
#pragma unroll
    for (int k = 0; k < 3; ++k) v[k] = src[tid + 256 * k];
    float s = v[0] + v[1] + v[2];
    red[tid] = s; __syncthreads();
    for (int off = 128; off > 0; off >>= 1) {
        if (tid < off) red[tid] += red[tid + off];
        __syncthreads();
    }
    float mu = red[0] / (float)CC; __syncthreads();
    float sq = 0.f;
#pragma unroll
    for (int k = 0; k < 3; ++k) { float d = v[k] - mu; sq += d * d; }
    red[tid] = sq; __syncthreads();
    for (int off = 128; off > 0; off >>= 1) {
        if (tid < off) red[tid] += red[tid + off];
        __syncthreads();
    }
    float rs = rsqrtf(red[0] / (float)CC + LN_EPS);
    __bf16* dst = out + (size_t)row * CC;
#pragma unroll
    for (int k = 0; k < 3; ++k) {
        int c = tid + 256 * k;
        dst[c] = (__bf16)((v[k] - mu) * rs * g[c] + b[c]);
    }
}

// ---------------- tiled WMMA GEMM, double-buffered LDS ----------------
// EPI 0: out=bf16 (QKV)  1: out=f32 resid+ (proj->x2)
// EPI 2: GELU -> bf16 (fc1)  3: out += gate * (resid + v) (fc2 combine)
#define BM 128
#define BN 128
#define BK 64

// A tile: 128x64 bf16. thread: row=tid>>1, seg=tid&1 (32 bf16 = 4 uint4)
__device__ __forceinline__ void load_A_regs(const __bf16* A, int Kdim, int row0, int k0,
                                            int ar, int aseg, uint4* r4) {
    const uint4* s4 = reinterpret_cast<const uint4*>(
        A + (size_t)(row0 + ar) * Kdim + k0 + aseg * 32);
#pragma unroll
    for (int j = 0; j < 4; ++j) r4[j] = s4[j];
}
__device__ __forceinline__ void store_A_lds(__bf16 (*As)[BK + 8], int ar, int aseg,
                                            const uint4* r4) {
    uint4* d4 = reinterpret_cast<uint4*>(&As[ar][aseg * 32]);
#pragma unroll
    for (int j = 0; j < 4; ++j) d4[j] = r4[j];
}
// W tile: 64x128 f32. thread: kq=tid>>4 (4 K-rows), g=tid&15 (8 N-cols)
__device__ __forceinline__ void load_W_regs(const float* W, int N, int col0, int k0,
                                            int kq, int g, float4* w4) {
#pragma unroll
    for (int kr = 0; kr < 4; ++kr)
#pragma unroll
        for (int nj = 0; nj < 2; ++nj)
            w4[kr * 2 + nj] = *reinterpret_cast<const float4*>(
                W + (size_t)(k0 + kq * 4 + kr) * N + col0 + g * 8 + nj * 4);
}
// pack 4 consecutive K per column -> one ds_store_b64 into Bs[n][k]
__device__ __forceinline__ void store_W_lds(__bf16 (*Bs)[BK + 8], int kq, int g,
                                            const float4* w4) {
#pragma unroll
    for (int c = 0; c < 8; ++c) {
        Pack4 pk;
#pragma unroll
        for (int kr = 0; kr < 4; ++kr) {
            F4 fv; fv.v = w4[kr * 2 + (c >> 2)];
            pk.h[kr] = (__bf16)fv.f[c & 3];
        }
        *reinterpret_cast<uint2*>(&Bs[g * 8 + c][kq * 4]) = pk.u;
    }
}

template <int EPI>
__global__ __launch_bounds__(256)
void gemm_bf16_kernel(const __bf16* __restrict__ A, const float* __restrict__ W,
                      const float* __restrict__ bias, int Kdim, int N,
                      const float* __restrict__ resid, float* __restrict__ outf,
                      __bf16* __restrict__ outb, const float* __restrict__ gate,
                      int expert) {
    __shared__ __align__(16) __bf16 As[2][BM][BK + 8];
    __shared__ __align__(16) __bf16 Bs[2][BN][BK + 8];   // [n][k]
    const int tid  = threadIdx.x;
    const int lane = tid & 31;
    const int wave = tid >> 5;
    const int wm   = wave >> 1;       // 0..3 -> 32 rows each
    const int wn   = wave & 1;        // 0..1 -> 64 cols each
    const int row0 = blockIdx.x * BM;
    const int col0 = blockIdx.y * BN;
    const int ar = tid >> 1, aseg = tid & 1;
    const int kq = tid >> 4, wg   = tid & 15;

    v8f acc[2][4];
    const v8f zf = {};
#pragma unroll
    for (int i = 0; i < 2; ++i)
#pragma unroll
        for (int j = 0; j < 4; ++j) acc[i][j] = zf;

    // prologue: stage tile 0 into buffer 0
    {
        uint4  a4[4];
        float4 w4[8];
        load_A_regs(A, Kdim, row0, 0, ar, aseg, a4);
        load_W_regs(W, N, col0, 0, kq, wg, w4);
        store_A_lds(As[0], ar, aseg, a4);
        store_W_lds(Bs[0], kq, wg, w4);
    }
    __syncthreads();

    const int nk = Kdim / BK;
    for (int kt = 0; kt < nk; ++kt) {
        const int cur = kt & 1, nxt = cur ^ 1;
        const bool pf = (kt + 1 < nk);
        uint4  a4[4];
        float4 w4[8];
        if (pf) {   // issue next tile's global loads before the math
            load_A_regs(A, Kdim, row0, (kt + 1) * BK, ar, aseg, a4);
            load_W_regs(W, N, col0, (kt + 1) * BK, kq, wg, w4);
        }
#pragma unroll
        for (int kk = 0; kk < BK / 32; ++kk) {
            v16bf a[2], b[4];
#pragma unroll
            for (int i = 0; i < 2; ++i)
                a[i] = load_a_frag(&As[cur][wm * 32 + i * 16][0], BK + 8, lane, kk * 32);
#pragma unroll
            for (int j = 0; j < 4; ++j)
                b[j] = load_b_frag(&Bs[cur][wn * 64 + j * 16][0], BK + 8, lane, kk * 32);
#pragma unroll
            for (int i = 0; i < 2; ++i)
#pragma unroll
                for (int j = 0; j < 4; ++j)
                    acc[i][j] = wmma_bf16(a[i], b[j], acc[i][j]);
        }
        if (pf) {
            store_A_lds(As[nxt], ar, aseg, a4);
            store_W_lds(Bs[nxt], kq, wg, w4);
        }
        __syncthreads();
    }

    // epilogue (C/D layout: VGPR r -> M = r + (lane>=16 ? 8:0), N = lane%16)
    const int rbase = row0 + wm * 32 + ((lane >= 16) ? 8 : 0);
    const int cbase = col0 + wn * 64 + (lane & 15);
#pragma unroll
    for (int i = 0; i < 2; ++i) {
#pragma unroll
        for (int j = 0; j < 4; ++j) {
            const int colj = cbase + j * 16;
            const float bv = bias[colj];
#pragma unroll
            for (int r = 0; r < 8; ++r) {
                const int row = rbase + i * 16 + r;
                const size_t idx = (size_t)row * N + colj;
                float v = acc[i][j][r] + bv;
                if (EPI == 0) {
                    outb[idx] = (__bf16)v;
                } else if (EPI == 1) {
                    outf[idx] = resid[idx] + v;
                } else if (EPI == 2) {
                    float ge = 0.5f * v * (1.f + erff(v * 0.7071067811865476f));
                    outb[idx] = (__bf16)ge;
                } else {
                    float gv = gate[(size_t)row * EE + expert];
                    outf[idx] += gv * (resid[idx] + v);
                }
            }
        }
    }
}

// ---------------- flash attention (per expert), bf16 WMMA ----------------
__global__ __launch_bounds__(256)
void attn_kernel(const __bf16* __restrict__ qkv, __bf16* __restrict__ attn_out) {
    const int qt = blockIdx.x;            // query tile 0..7
    const int bh = blockIdx.y;            // 0..B*H-1
    const int bb = bh / HH, hh = bh % HH;
    const int tid = threadIdx.x, lane = tid & 31, wave = tid >> 5;
    const int wm = wave >> 1, wn = wave & 1;

    __shared__ __align__(16) __bf16 Qs[64][72];
    __shared__ __align__(16) __bf16 Ks[64][72];       // [key][d] : B for Q@K^T
    __shared__ __align__(16) __bf16 Vt[64][72];       // [d][key] : B for P@V
    __shared__ __align__(16) __bf16 Ps[64][72];       // [q][key] : A for P@V
    __shared__ float Sc[64][68];
    __shared__ float mrow[64], lrow[64], arow[64];

    const size_t tokbase = (size_t)bb * SS;
    const int qbase = qt * 64;
    const int ldq = 3 * CC;
    const int lr = tid >> 2, lseg = tid & 3;
    const int vk = (tid & 31) * 2, vd = tid >> 5;     // V transpose loader

    {   // load Q tile once
        const uint4* s4 = reinterpret_cast<const uint4*>(
            qkv + (tokbase + qbase + lr) * ldq + hh * HD + lseg * 16);
        *reinterpret_cast<uint4*>(&Qs[lr][lseg * 16])     = s4[0];
        *reinterpret_cast<uint4*>(&Qs[lr][lseg * 16 + 8]) = s4[1];
    }
    if (tid < 64) { mrow[tid] = -1e30f; lrow[tid] = 0.f; }
    v8f o[2]; const v8f zf = {};
    o[0] = zf; o[1] = zf;
    __syncthreads();

    for (int kt = 0; kt < SS / 64; ++kt) {
        const int kb = kt * 64;
        {   // K tile
            const uint4* ks4 = reinterpret_cast<const uint4*>(
                qkv + (tokbase + kb + lr) * ldq + CC + hh * HD + lseg * 16);
            *reinterpret_cast<uint4*>(&Ks[lr][lseg * 16])     = ks4[0];
            *reinterpret_cast<uint4*>(&Ks[lr][lseg * 16 + 8]) = ks4[1];
            // V tile transposed: 2 keys per thread, pack pairs -> ds_store_b32
            Pack8 pa, pb;
            pa.u = *reinterpret_cast<const uint4*>(
                qkv + (tokbase + kb + vk) * ldq + 2 * CC + hh * HD + vd * 8);
            pb.u = *reinterpret_cast<const uint4*>(
                qkv + (tokbase + kb + vk + 1) * ldq + 2 * CC + hh * HD + vd * 8);
#pragma unroll
            for (int j = 0; j < 8; ++j) {
                Pack2 pk; pk.h[0] = pa.h[j]; pk.h[1] = pb.h[j];
                *reinterpret_cast<unsigned int*>(&Vt[vd * 8 + j][vk]) = pk.u;
            }
        }
        __syncthreads();

        // scores = Q @ K^T  (x sqrt(HD) per reference)
        v8f s[2]; s[0] = zf; s[1] = zf;
#pragma unroll
        for (int kk = 0; kk < 2; ++kk) {
            v16bf a = load_a_frag(&Qs[wm * 16][0], 72, lane, kk * 32);
#pragma unroll
            for (int j = 0; j < 2; ++j) {
                v16bf b = load_b_frag(&Ks[wn * 32 + j * 16][0], 72, lane, kk * 32);
                s[j] = wmma_bf16(a, b, s[j]);
            }
        }
        {
            const int r0 = wm * 16 + ((lane >= 16) ? 8 : 0);
            const int cc = wn * 32 + (lane & 15);
#pragma unroll
            for (int j = 0; j < 2; ++j)
#pragma unroll
                for (int r = 0; r < 8; ++r)
                    Sc[r0 + r][cc + j * 16] = s[j][r] * 8.0f;   // sqrt(64)
        }
        __syncthreads();

        // online softmax per row
        if (tid < 64) {
            float mold = mrow[tid], mx = mold;
            for (int j = 0; j < 64; ++j) mx = fmaxf(mx, Sc[tid][j]);
            float al = __expf(mold - mx), ssum = 0.f;
            for (int j = 0; j < 64; ++j) {
                float p = __expf(Sc[tid][j] - mx);
                Ps[tid][j] = (__bf16)p;
                ssum += p;
            }
            mrow[tid] = mx;
            lrow[tid] = lrow[tid] * al + ssum;
            arow[tid] = al;
        }
        __syncthreads();

        // rescale O, accumulate P @ V
        {
            const int r0 = wm * 16 + ((lane >= 16) ? 8 : 0);
#pragma unroll
            for (int j = 0; j < 2; ++j)
#pragma unroll
                for (int r = 0; r < 8; ++r) o[j][r] *= arow[r0 + r];
        }
#pragma unroll
        for (int kk = 0; kk < 2; ++kk) {
            v16bf a = load_a_frag(&Ps[wm * 16][0], 72, lane, kk * 32);
#pragma unroll
            for (int j = 0; j < 2; ++j) {
                v16bf b = load_b_frag(&Vt[wn * 32 + j * 16][0], 72, lane, kk * 32);
                o[j] = wmma_bf16(a, b, o[j]);
            }
        }
        __syncthreads();
    }

    {   // normalize, write bf16
        const int r0 = wm * 16 + ((lane >= 16) ? 8 : 0);
        const int c0 = hh * HD + wn * 32 + (lane & 15);
#pragma unroll
        for (int j = 0; j < 2; ++j)
#pragma unroll
            for (int r = 0; r < 8; ++r) {
                const int q = qbase + r0 + r;
                float val = o[j][r] / lrow[r0 + r];
                attn_out[(tokbase + q) * CC + c0 + j * 16] = (__bf16)val;
            }
    }
}

// ---------------- host launcher ----------------
extern "C" void kernel_launch(void* const* d_in, const int* in_sizes, int n_in,
                              void* d_out, int out_size, void* d_ws, size_t ws_size,
                              hipStream_t stream) {
    const float* x      = (const float*)d_in[0];
    const float* et     = (const float*)d_in[1];
    const float* ln1_g  = (const float*)d_in[2];
    const float* ln1_b  = (const float*)d_in[3];
    const float* qkv_w  = (const float*)d_in[4];
    const float* qkv_b  = (const float*)d_in[5];
    const float* proj_w = (const float*)d_in[6];
    const float* proj_b = (const float*)d_in[7];
    const float* ln2_g  = (const float*)d_in[8];
    const float* ln2_b  = (const float*)d_in[9];
    const float* fc1_w  = (const float*)d_in[10];
    const float* fc1_b  = (const float*)d_in[11];
    const float* fc2_w  = (const float*)d_in[12];
    const float* fc2_b  = (const float*)d_in[13];
    float* out = (float*)d_out;

    char* ws = (char*)d_ws;
    float*  gate  = (float*)(ws + 0);                             // 4096*8 f32
    float*  stats = (float*)(ws + 131072);                        // 16 f32
    __bf16* h1    = (__bf16*)(ws + 131328);                       // 4096*768
    __bf16* qkvb  = (__bf16*)(ws + 6422784);                      // 4096*2304
    __bf16* attnb = (__bf16*)(ws + 25297152);                     // 4096*768
    float*  x2    = (float*)(ws + 31588608);                      // 4096*768 f32
    __bf16* h2    = (__bf16*)(ws + 44171520);                     // 4096*768
    __bf16* actb  = (__bf16*)(ws + 50462976);                     // 4096*3072

    init_kernel<<<(TT * CC) / 256, 256, 0, stream>>>(x, out, stats);
    router_kernel<<<TT / 8, 256, 0, stream>>>(x, et, gate, stats);

    for (int e = 0; e < EE; ++e) {
        ln_bf16_kernel<<<TT, 256, 0, stream>>>(x, ln1_g + e * CC, ln1_b + e * CC, h1);
        gemm_bf16_kernel<0><<<dim3(TT / BM, (3 * CC) / BN), 256, 0, stream>>>(
            h1, qkv_w + (size_t)e * CC * 3 * CC, qkv_b + (size_t)e * 3 * CC,
            CC, 3 * CC, nullptr, nullptr, qkvb, nullptr, e);
        attn_kernel<<<dim3(SS / 64, BB * HH), 256, 0, stream>>>(qkvb, attnb);
        gemm_bf16_kernel<1><<<dim3(TT / BM, CC / BN), 256, 0, stream>>>(
            attnb, proj_w + (size_t)e * CC * CC, proj_b + (size_t)e * CC,
            CC, CC, x, x2, nullptr, nullptr, e);
        ln_bf16_kernel<<<TT, 256, 0, stream>>>(x2, ln2_g + e * CC, ln2_b + e * CC, h2);
        gemm_bf16_kernel<2><<<dim3(TT / BM, MLPH / BN), 256, 0, stream>>>(
            h2, fc1_w + (size_t)e * CC * MLPH, fc1_b + (size_t)e * MLPH,
            CC, MLPH, nullptr, nullptr, actb, nullptr, e);
        gemm_bf16_kernel<3><<<dim3(TT / BM, CC / BN), 256, 0, stream>>>(
            actb, fc2_w + (size_t)e * MLPH * CC, fc2_b + (size_t)e * CC,
            MLPH, CC, x2, out, nullptr, gate, e);
    }
    loss_kernel<<<1, 256, 0, stream>>>(et, stats, out + (size_t)TT * CC);
}
